// mLSTMBlock_67061619359783
// MI455X (gfx1250) — matrix operand order repaired and checked by reading, not compiled, not run on hardware
//
#include <hip/hip_runtime.h>
#include <hip/hip_bf16.h>
#include <math.h>

// ---------------------------------------------------------------------------
// mLSTM block for MI455X (gfx1250): bf16 WMMA GEMMs with async-to-LDS double
// buffering, pre-transposed bf16 weights, fused streaming kernels.
// ---------------------------------------------------------------------------
#define B_     4096
#define IN_    2048
#define HID_   4096
#define PROJ_  4096
#define H_     8
#define BH_    512
#define BI_    512

typedef __attribute__((ext_vector_type(16))) __bf16 v16bf;
typedef __attribute__((ext_vector_type(8)))  float  v8f;
typedef int v4i_gcc __attribute__((vector_size(16)));   // matches builtin proto

typedef union { uint4 u[2]; v16bf v; } frag16_t;

// --- gfx1250 async global->LDS copy path (ASYNCcnt), with safe fallback -----
#if __has_builtin(__builtin_amdgcn_global_load_async_to_lds_b128) && \
    __has_builtin(__builtin_amdgcn_s_wait_asynccnt)
#define USE_ASYNC_LDS 1
#else
#define USE_ASYNC_LDS 0
#endif

__device__ __forceinline__ void cpAsync16(__bf16* ldsDst, const __bf16* gSrc) {
#if USE_ASYNC_LDS
    __builtin_amdgcn_global_load_async_to_lds_b128(
        (__attribute__((address_space(1))) v4i_gcc*)(gSrc),
        (__attribute__((address_space(3))) v4i_gcc*)(ldsDst),
        0, 0);
#else
    *(uint4*)ldsDst = *(const uint4*)gSrc;
#endif
}

__device__ __forceinline__ void asyncWait() {
#if USE_ASYNC_LDS
    __builtin_amdgcn_s_wait_asynccnt(0);
#endif
}

// ---------------------------------------------------------------------------
// helpers: wave32 reductions
// ---------------------------------------------------------------------------
__device__ __forceinline__ float warpRed(float v) {
#pragma unroll
    for (int o = 16; o > 0; o >>= 1) v += __shfl_down(v, o, 32);
    return v;
}

__device__ __forceinline__ float blockRed256(float v, float* scr) {
    const int lane = threadIdx.x & 31;
    const int w    = threadIdx.x >> 5;
    v = warpRed(v);
    __syncthreads();
    if (lane == 0) scr[w] = v;
    __syncthreads();
    float r = scr[0];
#pragma unroll
    for (int i = 1; i < 8; ++i) r += scr[i];
    return r;
}

// ---------------------------------------------------------------------------
// fp32[K,N] -> bf16[N,K] transposing converter (32x32 tiles through LDS)
// blockIdx.z batches per-head weights via element strides.
// ---------------------------------------------------------------------------
__global__ __launch_bounds__(256) void k_f2bf_tr(const float* __restrict__ in,
                                                 __bf16* __restrict__ out,
                                                 int Kdim, int Ndim,
                                                 long long zIn, long long zOut) {
    __shared__ float tile[32][33];
    const float* src = in  + (size_t)blockIdx.z * (size_t)zIn;
    __bf16*      dst = out + (size_t)blockIdx.z * (size_t)zOut;
    const int k0 = blockIdx.y << 5;
    const int n0 = blockIdx.x << 5;
    const int t  = threadIdx.x;
    const int r  = t >> 3;         // 0..31
    const int c4 = (t & 7) << 2;   // 0..28
    const float4 f = *(const float4*)(src + (size_t)(k0 + r) * Ndim + n0 + c4);
    tile[r][c4 + 0] = f.x;
    tile[r][c4 + 1] = f.y;
    tile[r][c4 + 2] = f.z;
    tile[r][c4 + 3] = f.w;
    __syncthreads();
    union { __bf16 b[4]; uint2 u; } o;
#pragma unroll
    for (int i = 0; i < 4; ++i) o.b[i] = (__bf16)tile[c4 + i][r];
    *(uint2*)(dst + (size_t)(n0 + r) * Kdim + k0 + c4) = o.u;
}

// ---------------------------------------------------------------------------
// row LayerNorm over IN_ (2048) -> bf16
// ---------------------------------------------------------------------------
__global__ __launch_bounds__(256) void k_layernorm(const float* __restrict__ x,
                                                   const float* __restrict__ w,
                                                   const float* __restrict__ b,
                                                   __bf16* __restrict__ out) {
    __shared__ float sx[IN_];
    __shared__ float scr[8];
    const int row = blockIdx.x;
    const float* xr = x + (size_t)row * IN_;
    float s = 0.f, ss = 0.f;
    for (int j = threadIdx.x; j < IN_; j += 256) {
        float v = xr[j];
        sx[j] = v;
        s  += v;
        ss += v * v;
    }
    const float S  = blockRed256(s,  scr);
    const float SS = blockRed256(ss, scr);
    const float mu   = S * (1.f / IN_);
    const float var  = SS * (1.f / IN_) - mu * mu;
    const float rinv = rsqrtf(var + 1e-5f);
    __syncthreads();
    __bf16* o = out + (size_t)row * IN_;
    for (int j = threadIdx.x; j < IN_; j += 256)
        o[j] = (__bf16)((sx[j] - mu) * rinv * w[j] + b[j]);
}

// ---------------------------------------------------------------------------
// bf16 WMMA GEMM: C[M,N] = A[M,K] * Wt[N,K]^T + bias (+ residual)
//   256 threads = 8 waves, 128x128 tile, K-step 32, double-buffered LDS with
//   async global->LDS copies overlapping WMMA compute of the previous tile.
//   Wave layout: 4x2 grid of 32x64 sub-tiles => 2x4 accumulators of 16x16.
//   blockIdx.z batches the 8-head block-diagonal GEMMs via element strides.
// ---------------------------------------------------------------------------
__global__ __launch_bounds__(256) void k_gemm_bf16(
        const __bf16* __restrict__ A0, const __bf16* __restrict__ Wt0,
        const float* __restrict__ bias0, const float* __restrict__ resid,
        float* __restrict__ C0,
        int M, int N, int K, int lda, int ldwt, int ldc,
        long long aZ, long long wZ, long long cZ, int bZ) {
    const __bf16* A    = A0    + (size_t)blockIdx.z * (size_t)aZ;
    const __bf16* Wt   = Wt0   + (size_t)blockIdx.z * (size_t)wZ;
    float*        C    = C0    + (size_t)blockIdx.z * (size_t)cZ;
    const float*  bias = bias0 + (size_t)blockIdx.z * (size_t)bZ;

    const int m0 = blockIdx.y << 7;
    const int n0 = blockIdx.x << 7;

    __shared__ __bf16 As[2][128 * 32];   // [row][k], row stride 32
    __shared__ __bf16 Bs[2][128 * 32];   // [n][k],  row stride 32 (Wt layout)

    const int t     = threadIdx.x;
    const int wid   = t >> 5;
    const int lane  = t & 31;
    const int waveM = (wid & 3) << 5;   // 0,32,64,96
    const int waveN = (wid >> 2) << 6;  // 0,64
    const int half  = lane >> 4;        // lane-group per ISA 16-bit layouts
    const int lrow  = lane & 15;

    // staging assignment: thread t copies row sr, 2x16B chunks at sk, sk+8
    const int sr = t >> 1;             // 0..127
    const int sk = (t & 1) << 4;       // 0 or 16

    v8f acc[2][4];
#pragma unroll
    for (int mi = 0; mi < 2; ++mi)
#pragma unroll
        for (int ni = 0; ni < 4; ++ni)
#pragma unroll
            for (int j = 0; j < 8; ++j) acc[mi][ni][j] = 0.f;

    auto stage = [&](int buf, int k0) {
        const __bf16* ga = A  + (size_t)(m0 + sr) * lda  + k0 + sk;
        const __bf16* gb = Wt + (size_t)(n0 + sr) * ldwt + k0 + sk;
        cpAsync16(&As[buf][sr * 32 + sk],     ga);
        cpAsync16(&As[buf][sr * 32 + sk + 8], ga + 8);
        cpAsync16(&Bs[buf][sr * 32 + sk],     gb);
        cpAsync16(&Bs[buf][sr * 32 + sk + 8], gb + 8);
#if !USE_ASYNC_LDS
        if (k0 + 32 < K) {
            __builtin_prefetch(ga + 32, 0, 0);
            __builtin_prefetch(gb + 32, 0, 0);
        }
#endif
    };

    stage(0, 0);
    int buf = 0;
    for (int k0 = 0; k0 < K; k0 += 32) {
        asyncWait();                 // my copies for As/Bs[buf] complete
        __syncthreads();             // everyone's copies complete
        if (k0 + 32 < K) stage(buf ^ 1, k0 + 32);   // overlap with compute

        frag16_t a[2], b[4];
#pragma unroll
        for (int mi = 0; mi < 2; ++mi) {
            const int r = waveM + (mi << 4) + lrow;
            a[mi].u[0] = *(const uint4*)&As[buf][r * 32 + half * 16];
            a[mi].u[1] = *(const uint4*)&As[buf][r * 32 + half * 16 + 8];
        }
#pragma unroll
        for (int ni = 0; ni < 4; ++ni) {
            const int n = waveN + (ni << 4) + lrow;
            b[ni].u[0] = *(const uint4*)&Bs[buf][n * 32 + half * 16];
            b[ni].u[1] = *(const uint4*)&Bs[buf][n * 32 + half * 16 + 8];
        }
#pragma unroll
        for (int mi = 0; mi < 2; ++mi)
#pragma unroll
            for (int ni = 0; ni < 4; ++ni)
                acc[mi][ni] = __builtin_amdgcn_wmma_f32_16x16x32_bf16(
                    false, a[mi].v, false, b[ni].v,
                    (short)0, acc[mi][ni], false, false);
        buf ^= 1;
    }

    // epilogue: C layout = lane(L%16)->N, (half*8 + j)->M
#pragma unroll
    for (int mi = 0; mi < 2; ++mi)
#pragma unroll
        for (int ni = 0; ni < 4; ++ni) {
            const int col = n0 + waveN + (ni << 4) + lrow;
            const float bv = bias[col];
#pragma unroll
            for (int j = 0; j < 8; ++j) {
                const int row = m0 + waveM + (mi << 4) + half * 8 + j;
                const size_t ci = (size_t)row * ldc + col;
                float v = acc[mi][ni][j] + bv;
                if (resid) v += resid[ci];
                C[ci] = v;
            }
        }
}

// ---------------------------------------------------------------------------
// causal conv (K=4) over the feature dim + SiLU; also emits bf16 x_left
// ---------------------------------------------------------------------------
__global__ __launch_bounds__(256) void k_conv_silu(const float* __restrict__ xl,
                                                   const float* __restrict__ cw,
                                                   const float* __restrict__ cb,
                                                   __bf16* __restrict__ xc,
                                                   __bf16* __restrict__ xlb) {
    const size_t i = (size_t)blockIdx.x * 256 + threadIdx.x;  // over B_*PROJ_
    const int l = (int)(i & (PROJ_ - 1));
    const float x0 = xl[i];
    float acc = cb[0] + cw[3] * x0;
    if (l >= 1) acc += cw[2] * xl[i - 1];
    if (l >= 2) acc += cw[1] * xl[i - 2];
    if (l >= 3) acc += cw[0] * xl[i - 3];
    const float sv = acc / (1.f + expf(-acc));  // silu
    xc[i]  = (__bf16)sv;
    xlb[i] = (__bf16)x0;
}

// ---------------------------------------------------------------------------
// fused mLSTM state update + GroupNorm + skip + SiLU gate
// ---------------------------------------------------------------------------
__global__ __launch_bounds__(256) void k_state(
        const float* __restrict__ q,  const float* __restrict__ kk,
        const float* __restrict__ v,
        const float* __restrict__ it, const float* __restrict__ ft,
        const float* __restrict__ op,
        const float* __restrict__ cprev, const float* __restrict__ nprev,
        const float* __restrict__ mprev,
        const float* __restrict__ xskip, const float* __restrict__ xright,
        const float* __restrict__ gw, const float* __restrict__ gb,
        float* __restrict__ out_h, float* __restrict__ out_c,
        float* __restrict__ out_n, float* __restrict__ out_m,
        __bf16* __restrict__ pd) {
    __shared__ float sh[HID_];
    __shared__ float scr[8];
    const int row = blockIdx.x;
    const size_t base = (size_t)row * HID_;
    const float kscale = 0.04419417382415922f;  // 1/sqrt(512)

    float part = 0.f;
    for (int j = threadIdx.x; j < HID_; j += 256) {
        const size_t idx = base + j;
        const float itv = it[idx];
        const float ftv = ft[idx];
        const float mp  = mprev[idx];
        const float mt  = fmaxf(ftv + mp, itv);
        const float iv  = expf(itv - mt);
        const float fv  = expf(ftv + mp - mt);
        const float kv  = kk[idx] * kscale;
        const float qv  = q[idx];
        const float ct  = fv * cprev[idx] + iv * (v[idx] * kv);
        const float nt  = fv * nprev[idx] + iv * kv;
        out_c[idx] = ct;
        out_n[idx] = nt;
        out_m[idx] = mt;
        const float s = nt * qv;
        part += s * s;
        const float oo = 1.f / (1.f + expf(-op[idx]));
        sh[j] = oo * ct * qv;   // h_t numerator
    }
    const float tot  = blockRed256(part, scr);
    const float rden = 1.f / (sqrtf(tot) + 1e-6f);
    __syncthreads();
    for (int j = threadIdx.x; j < HID_; j += 256) {
        const float h = sh[j] * rden;
        out_h[base + j] = h;
        sh[j] = h;
    }
    __syncthreads();

    for (int g = 0; g < H_; ++g) {   // GroupNorm: 8 groups of 512
        const int j0 = g * BH_ + threadIdx.x;
        const float h0 = sh[j0];
        const float h1 = sh[j0 + 256];
        const float S  = blockRed256(h0 + h1, scr);
        const float SS = blockRed256(h0 * h0 + h1 * h1, scr);
        const float mu  = S * (1.f / BH_);
        const float var = SS * (1.f / BH_) - mu * mu;
        const float rv  = rsqrtf(var + 1e-5f);
#pragma unroll
        for (int e = 0; e < 2; ++e) {
            const int j = j0 + e * 256;
            const size_t idx = base + j;
            const float rn = (sh[j] - mu) * rv;
            float val = rn * gw[j] + gb[j] + xskip[idx];
            const float xr = xright[idx];
            val *= xr / (1.f + expf(-xr));   // * silu(x_right)
            pd[idx] = (__bf16)val;
        }
    }
}

// ---------------------------------------------------------------------------
// host launcher
// ---------------------------------------------------------------------------
static inline size_t alignUp(size_t v, size_t a) { return (v + a - 1) & ~(a - 1); }

extern "C" void kernel_launch(void* const* d_in, const int* in_sizes, int n_in,
                              void* d_out, int out_size, void* d_ws, size_t ws_size,
                              hipStream_t stream) {
    (void)in_sizes; (void)n_in; (void)out_size; (void)ws_size;

    const float* x      = (const float*)d_in[0];
    const float* c_prev = (const float*)d_in[2];
    const float* n_prev = (const float*)d_in[3];
    const float* m_prev = (const float*)d_in[4];
    const float* ln_w   = (const float*)d_in[5];
    const float* ln_b   = (const float*)d_in[6];
    const float* upL_w  = (const float*)d_in[7];
    const float* upL_b  = (const float*)d_in[8];
    const float* upR_w  = (const float*)d_in[9];
    const float* upR_b  = (const float*)d_in[10];
    const float* skip_w = (const float*)d_in[11];
    const float* skip_b = (const float*)d_in[12];
    const float* down_w = (const float*)d_in[13];
    const float* down_b = (const float*)d_in[14];
    const float* conv_w = (const float*)d_in[15];
    const float* conv_b = (const float*)d_in[16];
    const float* wq_w   = (const float*)d_in[17];
    const float* wq_b   = (const float*)d_in[18];
    const float* wk_w   = (const float*)d_in[19];
    const float* wk_b   = (const float*)d_in[20];
    const float* wv_w   = (const float*)d_in[21];
    const float* wv_b   = (const float*)d_in[22];
    const float* wi_w   = (const float*)d_in[23];
    const float* wi_b   = (const float*)d_in[24];
    const float* wf_w   = (const float*)d_in[25];
    const float* wf_b   = (const float*)d_in[26];
    const float* wo_w   = (const float*)d_in[27];
    const float* wo_b   = (const float*)d_in[28];
    const float* gn_w   = (const float*)d_in[29];
    const float* gn_b   = (const float*)d_in[30];

    float* out_y = (float*)d_out;
    float* out_h = out_y + (size_t)B_ * IN_;
    float* out_c = out_h + (size_t)B_ * HID_;
    float* out_n = out_c + (size_t)B_ * HID_;
    float* out_m = out_n + (size_t)B_ * HID_;

    char* ws = (char*)d_ws;
    size_t off = 0;
    auto alloc = [&](size_t bytes) -> void* {
        off = alignUp(off, 256);
        void* p = ws + off;
        off += bytes;
        return p;
    };
    const size_t nBI  = (size_t)B_ * IN_;
    const size_t nBP  = (size_t)B_ * PROJ_;
    const size_t nBH  = (size_t)B_ * HID_;
    const size_t nUPL = (size_t)IN_ * PROJ_;
    const size_t nUPR = (size_t)IN_ * HID_;
    const size_t nSQ  = (size_t)PROJ_ * HID_;
    const size_t nDWN = (size_t)HID_ * IN_;
    const size_t nBD  = (size_t)H_ * BI_ * BH_;

    __bf16* xn_bf  = (__bf16*)alloc(nBI * 2);
    float*  xl_f   = (float*)alloc(nBP * 4);
    __bf16* xl_bf  = (__bf16*)alloc(nBP * 2);
    __bf16* xc_bf  = (__bf16*)alloc(nBP * 2);
    float*  xr_f   = (float*)alloc(nBH * 4);
    float*  skip_f = (float*)alloc(nBH * 4);
    float*  q_f    = (float*)alloc(nBH * 4);
    float*  k_f    = (float*)alloc(nBH * 4);
    float*  v_f    = (float*)alloc(nBH * 4);
    float*  it_f   = (float*)alloc(nBH * 4);
    float*  ft_f   = (float*)alloc(nBH * 4);
    float*  op_f   = (float*)alloc(nBH * 4);
    __bf16* pd_bf  = (__bf16*)alloc(nBH * 2);
    // transposed bf16 weights: Wt[N,K]
    __bf16* upL_bf  = (__bf16*)alloc(nUPL * 2);
    __bf16* upR_bf  = (__bf16*)alloc(nUPR * 2);
    __bf16* skip_bf = (__bf16*)alloc(nSQ * 2);
    __bf16* down_bf = (__bf16*)alloc(nDWN * 2);
    __bf16* wq_bf   = (__bf16*)alloc(nBD * 2);
    __bf16* wk_bf   = (__bf16*)alloc(nBD * 2);
    __bf16* wv_bf   = (__bf16*)alloc(nBD * 2);
    __bf16* wi_bf   = (__bf16*)alloc(nSQ * 2);
    __bf16* wf_bf   = (__bf16*)alloc(nSQ * 2);
    __bf16* wo_bf   = (__bf16*)alloc(nSQ * 2);

    // convert + transpose weights: in [K,N] f32 -> out [N,K] bf16
    auto cvtT = [&](const float* src, __bf16* dst, int Kd, int Nd, int Z,
                    long long zI, long long zO) {
        dim3 grid((unsigned)(Nd / 32), (unsigned)(Kd / 32), (unsigned)Z);
        k_f2bf_tr<<<grid, dim3(256), 0, stream>>>(src, dst, Kd, Nd, zI, zO);
    };
    cvtT(upL_w,  upL_bf,  IN_,   PROJ_, 1, 0, 0);
    cvtT(upR_w,  upR_bf,  IN_,   HID_,  1, 0, 0);
    cvtT(skip_w, skip_bf, PROJ_, HID_,  1, 0, 0);
    cvtT(down_w, down_bf, HID_,  IN_,   1, 0, 0);
    cvtT(wi_w,   wi_bf,   PROJ_, HID_,  1, 0, 0);
    cvtT(wf_w,   wf_bf,   PROJ_, HID_,  1, 0, 0);
    cvtT(wo_w,   wo_bf,   PROJ_, HID_,  1, 0, 0);
    cvtT(wq_w,   wq_bf,   BI_,   BH_,   H_, (long long)BI_ * BH_, (long long)BI_ * BH_);
    cvtT(wk_w,   wk_bf,   BI_,   BH_,   H_, (long long)BI_ * BH_, (long long)BI_ * BH_);
    cvtT(wv_w,   wv_bf,   BI_,   BH_,   H_, (long long)BI_ * BH_, (long long)BI_ * BH_);

    // 1) LayerNorm
    k_layernorm<<<dim3(B_), dim3(256), 0, stream>>>(x, ln_w, ln_b, xn_bf);

    auto gemm = [&](const __bf16* A, const __bf16* Wt, const float* bias,
                    const float* resid, float* C,
                    int M, int N, int K, int lda, int ldwt, int ldc,
                    int Z, long long aZ, long long wZ, long long cZ, int bZ) {
        dim3 grid((unsigned)(N / 128), (unsigned)(M / 128), (unsigned)Z);
        k_gemm_bf16<<<grid, dim3(256), 0, stream>>>(A, Wt, bias, resid, C,
                                                    M, N, K, lda, ldwt, ldc,
                                                    aZ, wZ, cZ, bZ);
    };

    // 2) up-projections
    gemm(xn_bf, upL_bf, upL_b, nullptr, xl_f, B_, PROJ_, IN_, IN_, IN_, PROJ_,
         1, 0, 0, 0, 0);
    gemm(xn_bf, upR_bf, upR_b, nullptr, xr_f, B_, HID_, IN_, IN_, IN_, HID_,
         1, 0, 0, 0, 0);

    // 3) causal conv + SiLU (also emits bf16 x_left)
    k_conv_silu<<<dim3((unsigned)(nBP / 256)), dim3(256), 0, stream>>>(
        xl_f, conv_w, conv_b, xc_bf, xl_bf);

    // 4) dense GEMMs off x_conv / x_left
    gemm(xc_bf, skip_bf, skip_b, nullptr, skip_f, B_, HID_, PROJ_, PROJ_, PROJ_, HID_,
         1, 0, 0, 0, 0);
    gemm(xc_bf, wi_bf, wi_b, nullptr, it_f, B_, HID_, PROJ_, PROJ_, PROJ_, HID_,
         1, 0, 0, 0, 0);
    gemm(xc_bf, wf_bf, wf_b, nullptr, ft_f, B_, HID_, PROJ_, PROJ_, PROJ_, HID_,
         1, 0, 0, 0, 0);
    gemm(xl_bf, wo_bf, wo_b, nullptr, op_f, B_, HID_, PROJ_, PROJ_, PROJ_, HID_,
         1, 0, 0, 0, 0);

    // 5) block-diagonal q/k/v: per-head 4096x512x512, batched over blockIdx.z
    gemm(xc_bf, wq_bf, wq_b, nullptr, q_f, B_, BH_, BI_, PROJ_, BI_, HID_,
         H_, BI_, (long long)BI_ * BH_, BH_, BH_);
    gemm(xc_bf, wk_bf, wk_b, nullptr, k_f, B_, BH_, BI_, PROJ_, BI_, HID_,
         H_, BI_, (long long)BI_ * BH_, BH_, BH_);
    gemm(xl_bf, wv_bf, wv_b, nullptr, v_f, B_, BH_, BI_, PROJ_, BI_, HID_,
         H_, BI_, (long long)BI_ * BH_, BH_, BH_);

    // 6) fused state update + GroupNorm + skip + SiLU gate
    k_state<<<dim3(B_), dim3(256), 0, stream>>>(
        q_f, k_f, v_f, it_f, ft_f, op_f, c_prev, n_prev, m_prev,
        skip_f, xr_f, gn_w, gn_b, out_h, out_c, out_n, out_m, pd_bf);

    // 7) down-projection with residual (+x) straight into d_out
    gemm(pd_bf, down_bf, down_b, x, out_y, B_, IN_, HID_, HID_, HID_, IN_,
         1, 0, 0, 0, 0);
}
